// LSTMModel_17411797418378
// MI455X (gfx1250) — compile-verified, matrix operand order checked by reading, not run on hardware
//
#include <hip/hip_runtime.h>

typedef __attribute__((ext_vector_type(16))) __bf16          v16bf;
typedef __attribute__((ext_vector_type(8)))  __bf16          bf16x8;
typedef __attribute__((ext_vector_type(8)))  float           v8f;
typedef __attribute__((ext_vector_type(8)))  unsigned short  u16x8;

#define HDIM   50
#define TILE_B 64
#define TFIX   512

__device__ __forceinline__ unsigned short f2bf(float f) {
    unsigned int u = __builtin_bit_cast(unsigned int, f);
    u += 0x7fffu + ((u >> 16) & 1u);          // round-to-nearest-even
    return (unsigned short)(u >> 16);
}
// fast activations: single v_exp_f32 + v_rcp_f32 (no IEEE-div fixup sequence)
__device__ __forceinline__ float fsig(float x)  {
    return __builtin_amdgcn_rcpf(1.f + __expf(-x));
}
__device__ __forceinline__ float ftanh(float x) {
    return __builtin_amdgcn_rcpf(1.f + __expf(-2.f * x)) * 2.f - 1.f;
}

// Load a 16x32 bf16 A-fragment from an LDS row-major [64 rows][64 k] bf16 tile.
// ISA 7.12.2 16-bit A layout: lane l -> row = l&15 (+16*mm), kbase = (l>>4)*8,
// element i -> K = kh*32 + kbase + (i&7) + (i>=8 ? 16 : 0)  => two contiguous 16B chunks.
__device__ __forceinline__ v16bf load_afrag(const unsigned short* base, int mm, int kh,
                                            int llo, int lhi) {
    int off = (mm * 16 + llo) * 64 + kh * 32 + lhi * 8;
    u16x8 lo = *(const u16x8*)(base + off);
    u16x8 hi = *(const u16x8*)(base + off + 16);
    bf16x8 l = __builtin_bit_cast(bf16x8, lo);
    bf16x8 h = __builtin_bit_cast(bf16x8, hi);
    return __builtin_shufflevector(l, h, 0,1,2,3,4,5,6,7,8,9,10,11,12,13,14,15);
}

#define WMMA_BF16(A, B, C) \
    __builtin_amdgcn_wmma_f32_16x16x32_bf16(false, (A), false, (B), (short)0, (C), false, false)

__global__ __launch_bounds__(128, 1)
void lstm2_fused_wmma(const float* __restrict__ x,
                      const float* __restrict__ Wih0, const float* __restrict__ Whh0,
                      const float* __restrict__ bih0, const float* __restrict__ bhh0,
                      const float* __restrict__ Wih1, const float* __restrict__ Whh1,
                      const float* __restrict__ bih1, const float* __restrict__ bhh1,
                      const float* __restrict__ fcW,  const float* __restrict__ fcb,
                      float* __restrict__ out, int Bn, int Tn)
{
    // 160 KB LDS: xT stage (128K) | h0 dbl-buf (16K) | h2 dbl-buf (16K); h2f aliases xT head.
    __shared__ __align__(16) unsigned char smem[(128 + 16 + 16) * 1024];
    float*          xT  = (float*)smem;                                 // [t][row^swiz] 64*512
    unsigned short* h0b = (unsigned short*)(smem + 128 * 1024);         // 2 x 64x64 bf16
    unsigned short* h2b = (unsigned short*)(smem + 144 * 1024);         // 2 x 64x64 bf16
    float*          h2f = (float*)smem;                                 // alias: xT[t<64] dead by then

    const int tid = threadIdx.x;
    const int w   = tid >> 5;       // wave id 0..3 : owns unit columns j = 16w..16w+15
    const int l   = tid & 31;
    const int lhi = l >> 4;
    const int llo = l & 15;
    const int j   = w * 16 + llo;   // padded unit column 0..63
    const int rowBase = blockIdx.x * TILE_B;
    const bool jv = (j < HDIM);

    // ---- Stage whole x tile into LDS (transposed, XOR-swizzled per 4 rows) ----
    for (int q = tid; q < TILE_B * TFIX / 4; q += 128) {
        int fidx = q * 4;
        int row  = fidx >> 9;            // /512
        int tt   = fidx & 511;
        const float4 v = *(const float4*)&x[(long)(rowBase + row) * Tn + tt];
        float vv[4] = { v.x, v.y, v.z, v.w };
        #pragma unroll
        for (int e = 0; e < 4; ++e) {
            int te = tt + e;
            xT[te * 64 + (row ^ ((te & 15) << 2))] = vv[e];
        }
    }

    // ---- Persistent weight B-fragments in registers: [mat][gate][khalf] ----
    v16bf wB[3][4][2];
    const float* mats[3] = { Whh0, Wih1, Whh1 };
    #pragma unroll
    for (int m = 0; m < 3; ++m) {
        const float* Wsrc = mats[m];
        #pragma unroll
        for (int g = 0; g < 4; ++g) {
            #pragma unroll
            for (int kh = 0; kh < 2; ++kh) {
                v16bf frag;
                #pragma unroll
                for (int i = 0; i < 16; ++i) {
                    int k = kh * 32 + lhi * 8 + (i & 7) + ((i >> 3) << 4);
                    float v = (jv && k < HDIM) ? Wsrc[(g * HDIM + j) * HDIM + k] : 0.f;
                    frag[i] = __builtin_bit_cast(__bf16, f2bf(v));
                }
                wB[m][g][kh] = frag;
            }
        }
    }

    float bias0[4], cf0[4], bias1[4];
    #pragma unroll
    for (int g = 0; g < 4; ++g) {
        bias0[g] = jv ? (bih0[g * HDIM + j] + bhh0[g * HDIM + j]) : 0.f;
        cf0[g]   = jv ? Wih0[g * HDIM + j] : 0.f;               // D=1 rank-1 input term
        bias1[g] = jv ? (bih1[g * HDIM + j] + bhh1[g * HDIM + j]) : 0.f;
    }

    v8f c0s[4], c1s[4];
    #pragma unroll
    for (int m = 0; m < 4; ++m) {
        c0s[m] = (v8f){0,0,0,0,0,0,0,0};
        c1s[m] = (v8f){0,0,0,0,0,0,0,0};
    }

    // zero initial h state (both parities)
    {
        unsigned int* z0 = (unsigned int*)h0b;
        unsigned int* z2 = (unsigned int*)h2b;
        for (int idx = tid; idx < 4096; idx += 128) { z0[idx] = 0u; z2[idx] = 0u; }
    }
    __syncthreads();

    for (int t = 0; t < Tn; ++t) {
        const int p = t & 1;
        const unsigned short* h0rd = h0b + p * 4096;
        unsigned short*       h0wr = h0b + (p ^ 1) * 4096;
        const unsigned short* h2rd = h2b + p * 4096;
        unsigned short*       h2wr = h2b + (p ^ 1) * 4096;
        const int swiz = (t & 15) << 2;
        const float* xrow = &xT[t * 64];

        // ---------------- layer 0 ----------------
        #pragma unroll
        for (int mm = 0; mm < 4; ++mm) {
            v16bf a0 = load_afrag(h0rd, mm, 0, llo, lhi);
            v16bf a1 = load_afrag(h0rd, mm, 1, llo, lhi);
            const int rb = mm * 16 + lhi * 8;
            const float4 xsA = *(const float4*)&xrow[(rb    ) ^ swiz];
            const float4 xsB = *(const float4*)&xrow[(rb + 4) ^ swiz];
            float xs[8] = { xsA.x, xsA.y, xsA.z, xsA.w, xsB.x, xsB.y, xsB.z, xsB.w };

            v8f acc[4];
            #pragma unroll
            for (int g = 0; g < 4; ++g) {
                #pragma unroll
                for (int r = 0; r < 8; ++r) acc[g][r] = xs[r] * cf0[g] + bias0[g];
                acc[g] = WMMA_BF16(a0, wB[0][g][0], acc[g]);
                acc[g] = WMMA_BF16(a1, wB[0][g][1], acc[g]);
            }
            v8f c = c0s[mm];
            #pragma unroll
            for (int r = 0; r < 8; ++r) {
                float ig = fsig(acc[0][r]);
                float fg = fsig(acc[1][r]);
                float gg = ftanh(acc[2][r]);
                float og = fsig(acc[3][r]);
                float cn = fg * c[r] + ig * gg;
                c[r] = cn;
                float hv = og * ftanh(cn);
                h0wr[(mm * 16 + r + lhi * 8) * 64 + j] = f2bf(hv);
            }
            c0s[mm] = c;
        }
        __syncthreads();

        // ---------------- layer 1 ----------------
        #pragma unroll
        for (int mm = 0; mm < 4; ++mm) {
            v16bf a10 = load_afrag(h0wr, mm, 0, llo, lhi);   // this step's h1
            v16bf a11 = load_afrag(h0wr, mm, 1, llo, lhi);
            v16bf a20 = load_afrag(h2rd, mm, 0, llo, lhi);   // previous h2
            v16bf a21 = load_afrag(h2rd, mm, 1, llo, lhi);

            v8f acc[4];
            #pragma unroll
            for (int g = 0; g < 4; ++g) {
                #pragma unroll
                for (int r = 0; r < 8; ++r) acc[g][r] = bias1[g];
                acc[g] = WMMA_BF16(a10, wB[1][g][0], acc[g]);
                acc[g] = WMMA_BF16(a11, wB[1][g][1], acc[g]);
                acc[g] = WMMA_BF16(a20, wB[2][g][0], acc[g]);
                acc[g] = WMMA_BF16(a21, wB[2][g][1], acc[g]);
            }
            v8f c = c1s[mm];
            #pragma unroll
            for (int r = 0; r < 8; ++r) {
                float ig = fsig(acc[0][r]);
                float fg = fsig(acc[1][r]);
                float gg = ftanh(acc[2][r]);
                float og = fsig(acc[3][r]);
                float cn = fg * c[r] + ig * gg;
                c[r] = cn;
                float hv = og * ftanh(cn);
                int row = mm * 16 + r + lhi * 8;
                h2wr[row * 64 + j] = f2bf(hv);
                if (t == Tn - 1) h2f[row * 64 + j] = hv;   // aliases xT[t<64]: long dead
            }
            c1s[mm] = c;
        }
        __syncthreads();
    }

    // ---------------- FC: out = h2_last @ fcW^T + fcb ----------------
    if (tid < TILE_B) {
        float s = fcb[0];
        #pragma unroll 1
        for (int jj = 0; jj < HDIM; ++jj) s += h2f[tid * 64 + jj] * fcW[jj];
        out[rowBase + tid] = s;
    }
}

extern "C" void kernel_launch(void* const* d_in, const int* in_sizes, int n_in,
                              void* d_out, int out_size, void* d_ws, size_t ws_size,
                              hipStream_t stream) {
    const float* x    = (const float*)d_in[0];
    const float* Wih0 = (const float*)d_in[1];
    const float* Whh0 = (const float*)d_in[2];
    const float* bih0 = (const float*)d_in[3];
    const float* bhh0 = (const float*)d_in[4];
    const float* Wih1 = (const float*)d_in[5];
    const float* Whh1 = (const float*)d_in[6];
    const float* bih1 = (const float*)d_in[7];
    const float* bhh1 = (const float*)d_in[8];
    const float* fcW  = (const float*)d_in[9];
    const float* fcb  = (const float*)d_in[10];
    float* out = (float*)d_out;

    const int T = TFIX;
    const int B = in_sizes[0] / T;           // D == 1
    dim3 grid(B / TILE_B), block(128);
    hipLaunchKernelGGL(lstm2_fused_wmma, grid, block, 0, stream,
                       x, Wih0, Whh0, bih0, bhh0, Wih1, Whh1, bih1, bhh1,
                       fcW, fcb, out, B, T);
}
